// AttentionBlock_9045201125436
// MI455X (gfx1250) — matrix operand order repaired and checked by reading, not compiled, not run on hardware
//
#include <hip/hip_runtime.h>
#include <hip/hip_bf16.h>
#include <stdint.h>

typedef __attribute__((ext_vector_type(16))) __bf16 v16bf;
typedef __attribute__((ext_vector_type(8)))  float  v8f;

#define BATCH 4
#define SEQ   4096
#define DV    128
#define DA    64

union Frag16 { v16bf v; unsigned u[8]; };

__device__ __forceinline__ v8f wmma_bf16(v16bf a, v16bf b, v8f c) {
  // 8 args: (neg_a, A, neg_b, B, c_mod, C, reuse_a, reuse_b)
  return __builtin_amdgcn_wmma_f32_16x16x32_bf16(false, a, false, b, (short)0, c,
                                                 false, false);
}

// ---- cross-lane reduction without DS: DPP8 xor1/2/4 + DPP16 row_mirror ----
constexpr int dpp8sel(int a, int b, int c, int d, int e, int f, int g, int h) {
  return a | (b << 3) | (c << 6) | (d << 9) | (e << 12) | (f << 15) |
         (g << 18) | (h << 21);
}
constexpr int SEL_XOR1 = dpp8sel(1, 0, 3, 2, 5, 4, 7, 6);
constexpr int SEL_XOR2 = dpp8sel(2, 3, 0, 1, 6, 7, 4, 5);
constexpr int SEL_XOR4 = dpp8sel(4, 5, 6, 7, 0, 1, 2, 3);

template <int SEL>
__device__ __forceinline__ float dpp8_f(float x) {
  return __builtin_bit_cast(
      float, __builtin_amdgcn_mov_dpp8(__builtin_bit_cast(int, x), SEL));
}
// DPP16 row_mirror (0x140): lane i reads lane 15-i within its row of 16.
__device__ __forceinline__ float dpp_mirror_f(float x) {
  return __builtin_bit_cast(
      float, __builtin_amdgcn_mov_dpp(__builtin_bit_cast(int, x), 0x140, 0xf,
                                      0xf, true));
}
// Reduce across the 16 lanes of each half-wave (rows of 16). After xor1/2/4
// each 8-lane group is uniform, so row_mirror legally combines the halves.
__device__ __forceinline__ float redmax16(float v) {
  v = fmaxf(v, dpp8_f<SEL_XOR1>(v));
  v = fmaxf(v, dpp8_f<SEL_XOR2>(v));
  v = fmaxf(v, dpp8_f<SEL_XOR4>(v));
  v = fmaxf(v, dpp_mirror_f(v));
  return v;
}
__device__ __forceinline__ float redsum16(float v) {
  v += dpp8_f<SEL_XOR1>(v);
  v += dpp8_f<SEL_XOR2>(v);
  v += dpp8_f<SEL_XOR4>(v);
  v += dpp_mirror_f(v);
  return v;
}

// -------------------------------------------------------------------------
// Phase 0: memory-bound fp32 -> bf16 conversion (4 elems/thread, b128 in,
// b64 out). Keeps all conversion VALU out of the GEMM inner loops.
// -------------------------------------------------------------------------
__global__ __launch_bounds__(256) void cvt_bf16_kernel(
    const float* __restrict__ src, __bf16* __restrict__ dst, int n)
{
  const int i = (blockIdx.x * 256 + threadIdx.x) * 4;
  if (i < n) {
    const float4 f = *reinterpret_cast<const float4*>(src + i);
    union { __bf16 h[4]; uint2 u; } t;
    t.h[0] = (__bf16)f.x;
    t.h[1] = (__bf16)f.y;
    t.h[2] = (__bf16)f.z;
    t.h[3] = (__bf16)f.w;
    *reinterpret_cast<uint2*>(dst + i) = t.u;
  }
}

// -------------------------------------------------------------------------
// Phase 1: QKV projections as a WMMA GEMM: Y[16384x256] = xb[16384x128]·Wb^T.
// All operands pre-converted bf16 -> fragments are plain aligned b32 loads.
// 8 waves/block; each wave owns 16 rows, sweeps 16 N-blocks (Q:0-3, K:4-7,
// V:8-15), K=128 as 4 k-steps -> 64 WMMAs per wave. All four B-fragments
// are materialized before the WMMA chain so the loads clause and overlap.
// Q pre-scaled by 1/sqrt(64); V scatter-stored transposed [B][128][SEQ].
// -------------------------------------------------------------------------
__global__ __launch_bounds__(256) void qkv_proj_wmma(
    const __bf16* __restrict__ xb,  const __bf16* __restrict__ Wqb,
    const __bf16* __restrict__ Wkb, const __bf16* __restrict__ Wvb,
    __bf16* __restrict__ Qb, __bf16* __restrict__ Kb, __bf16* __restrict__ Vt)
{
  const int tid  = threadIdx.x;
  const int w    = tid >> 5;
  const int lane = tid & 31;
  const int g    = lane >> 4;
  const int ln   = lane & 15;
  const int row0 = blockIdx.x * 128 + w * 16;   // global row = b*SEQ + s
  const int bb   = row0 >> 12;
  const int s0   = row0 & (SEQ - 1);

  // A-fragments: 16 rows x 128 dims = 4 k-steps of 32
  Frag16 xa[4];
  {
    const __bf16* Xrow = xb + (size_t)(row0 + ln) * DV;
#pragma unroll
    for (int s = 0; s < 4; ++s)
#pragma unroll
      for (int v = 0; v < 8; ++v) {
        const int k = 32 * s + 16 * g + 2 * v;
        xa[s].u[v] = *reinterpret_cast<const unsigned*>(Xrow + k);
      }
  }

  v8f zero = {};
#pragma unroll
  for (int nb = 0; nb < 16; ++nb) {
    // B-fragment source: W row = output column (y = x @ W^T)
    const __bf16* Wrow;
    if (nb < 4)       Wrow = Wqb + (size_t)(nb * 16 + ln) * DV;
    else if (nb < 8)  Wrow = Wkb + (size_t)((nb - 4) * 16 + ln) * DV;
    else              Wrow = Wvb + (size_t)((nb - 8) * 16 + ln) * DV;

    Frag16 wbf[4];
#pragma unroll
    for (int s = 0; s < 4; ++s)
#pragma unroll
      for (int v = 0; v < 8; ++v)
        wbf[s].u[v] =
            *reinterpret_cast<const unsigned*>(Wrow + 32 * s + 16 * g + 2 * v);

    v8f c = zero;
#pragma unroll
    for (int s = 0; s < 4; ++s) c = wmma_bf16(xa[s].v, wbf[s].v, c);

    if (nb < 8) {
      __bf16* dst = (nb < 4) ? Qb : Kb;
      const int   ncol = (nb & 3) * 16 + ln;
      const float sc   = (nb < 4) ? 0.125f : 1.0f;   // fold 1/sqrt(64) into Q
#pragma unroll
      for (int r = 0; r < 8; ++r)
        dst[(size_t)(row0 + 8 * g + r) * DA + ncol] = (__bf16)(c[r] * sc);
    } else {
      const int dim = (nb - 8) * 16 + ln;
#pragma unroll
      for (int r = 0; r < 8; ++r)
        Vt[((size_t)bb * DV + dim) * SEQ + (s0 + 8 * g + r)] = (__bf16)c[r];
    }
  }
}

// -------------------------------------------------------------------------
// Phase 2: flash attention. 8 waves/block, each wave owns 16 query rows,
// streams keys in 64-wide tiles with online softmax, O accumulated in fp32.
// Per tile: 8 QK^T WMMAs + 16 PV WMMAs; softmax reductions are DPP-only.
// -------------------------------------------------------------------------
__global__ __launch_bounds__(256) void attn_flash_kernel(
    const __bf16* __restrict__ Qb, const __bf16* __restrict__ Kb,
    const __bf16* __restrict__ Vt, float* __restrict__ out)
{
  __shared__ alignas(16) __bf16 pbuf[8][16 * 64];   // per-wave P staging (2 KB)

  const int tid  = threadIdx.x;
  const int w    = tid >> 5;       // wave id 0..7
  const int lane = tid & 31;
  const int g    = lane >> 4;      // lane half (0/1)
  const int ln   = lane & 15;
  const int b    = blockIdx.y;
  const int q0   = blockIdx.x * 128 + w * 16;

  const __bf16* __restrict__ Kbase = Kb + (size_t)b * SEQ * DA;
  const __bf16* __restrict__ Vbase = Vt + (size_t)b * DV * SEQ;

  // ---- Q A-fragments: 16x64 bf16 = two 16x32 frags (k-steps) -------------
  Frag16 qa[2];
  {
    const __bf16* Qrow = Qb + ((size_t)b * SEQ + q0 + ln) * DA;
#pragma unroll
    for (int s = 0; s < 2; ++s)
#pragma unroll
      for (int v = 0; v < 8; ++v) {
        const int k = 32 * s + 16 * g + 2 * v;
        qa[s].u[v] = *reinterpret_cast<const unsigned*>(Qrow + k);
      }
  }

  v8f zero = {};
  v8f o_acc[8];
#pragma unroll
  for (int d = 0; d < 8; ++d) o_acc[d] = zero;
  float m_i[8], l_i[8];
#pragma unroll
  for (int r = 0; r < 8; ++r) { m_i[r] = -1e30f; l_i[r] = 0.f; }

  for (int j = 0; j < SEQ; j += 64) {
    // ---- load all K B-fragments for the 64-key tile, then 8 WMMAs --------
    Frag16 kf[4][2];
#pragma unroll
    for (int c = 0; c < 4; ++c) {
      const __bf16* Krow = Kbase + (size_t)(j + 16 * c + ln) * DA;
#pragma unroll
      for (int v = 0; v < 8; ++v) {
        kf[c][0].u[v] = *reinterpret_cast<const unsigned*>(Krow + 16 * g + 2 * v);
        kf[c][1].u[v] =
            *reinterpret_cast<const unsigned*>(Krow + 32 + 16 * g + 2 * v);
      }
    }
    if (j + 64 < SEQ) {
      __builtin_prefetch(Kbase + (size_t)(j + 64 + lane) * DA, 0, 3);
      __builtin_prefetch(Kbase + (size_t)(j + 96 + lane) * DA, 0, 3);
    }

    v8f sc[4];
#pragma unroll
    for (int c = 0; c < 4; ++c) {
      v8f s = zero;
      s = wmma_bf16(qa[0].v, kf[c][0].v, s);
      s = wmma_bf16(qa[1].v, kf[c][1].v, s);
      sc[c] = s;
    }

    // ---- online softmax: C layout lane=col, reg r -> row 8g+r ------------
    float alpha[8];
#pragma unroll
    for (int r = 0; r < 8; ++r) {
      const float t4 = fmaxf(fmaxf(sc[0][r], sc[1][r]),
                             fmaxf(sc[2][r], sc[3][r]));
      const float mnew = fmaxf(m_i[r], redmax16(t4));
      alpha[r] = __expf(m_i[r] - mnew);
      const float p0 = __expf(sc[0][r] - mnew);
      const float p1 = __expf(sc[1][r] - mnew);
      const float p2 = __expf(sc[2][r] - mnew);
      const float p3 = __expf(sc[3][r] - mnew);
      l_i[r] = l_i[r] * alpha[r] + redsum16((p0 + p1) + (p2 + p3));
      m_i[r] = mnew;
      __bf16* pr = &pbuf[w][(8 * g + r) * 64 + ln];
      pr[0]  = (__bf16)p0;
      pr[16] = (__bf16)p1;
      pr[32] = (__bf16)p2;
      pr[48] = (__bf16)p3;
    }
    asm volatile("s_wait_dscnt 0" ::: "memory");  // wave-private LDS handoff

    // ---- reload P as two 16x32 A-fragments -------------------------------
    Frag16 pa[2];
#pragma unroll
    for (int t = 0; t < 2; ++t)
#pragma unroll
      for (int v = 0; v < 8; ++v) {
        const int k = 32 * t + 16 * g + 2 * v;
        pa[t].u[v] = *reinterpret_cast<const unsigned*>(&pbuf[w][ln * 64 + k]);
      }

    // ---- O = O*alpha + P V : 8 dim-blocks of 16 --------------------------
#pragma unroll
    for (int d = 0; d < 8; ++d) {
      Frag16 vb0, vb1;
      const __bf16* Vrow = Vbase + (size_t)(16 * d + ln) * SEQ + j;
#pragma unroll
      for (int v = 0; v < 8; ++v) {
        vb0.u[v] = *reinterpret_cast<const unsigned*>(Vrow + 16 * g + 2 * v);
        vb1.u[v] = *reinterpret_cast<const unsigned*>(Vrow + 32 + 16 * g + 2 * v);
      }
#pragma unroll
      for (int r = 0; r < 8; ++r) o_acc[d][r] *= alpha[r];
      o_acc[d] = wmma_bf16(pa[0].v, vb0.v, o_acc[d]);
      o_acc[d] = wmma_bf16(pa[1].v, vb1.v, o_acc[d]);
    }
  }

  // ---- epilogue: normalize and store fp32 --------------------------------
  float rl[8];
#pragma unroll
  for (int r = 0; r < 8; ++r) rl[r] = 1.0f / l_i[r];
#pragma unroll
  for (int d = 0; d < 8; ++d)
#pragma unroll
    for (int r = 0; r < 8; ++r) {
      const size_t q = (size_t)b * SEQ + q0 + 8 * g + r;
      out[q * DV + 16 * d + ln] = o_acc[d][r] * rl[r];
    }
}

extern "C" void kernel_launch(void* const* d_in, const int* in_sizes, int n_in,
                              void* d_out, int out_size, void* d_ws, size_t ws_size,
                              hipStream_t stream) {
  (void)in_sizes; (void)n_in; (void)out_size; (void)ws_size;
  const float* x  = (const float*)d_in[0];
  const float* Wq = (const float*)d_in[1];
  const float* Wk = (const float*)d_in[2];
  const float* Wv = (const float*)d_in[3];

  const size_t nX  = (size_t)BATCH * SEQ * DV;   // 2,097,152
  const size_t nWq = (size_t)DA * DV;            // 8,192
  const size_t nWk = (size_t)DA * DV;            // 8,192
  const size_t nWv = (size_t)DV * DV;            // 16,384

  __bf16* Qb  = (__bf16*)d_ws;                       // 2 MB
  __bf16* Kb  = Qb  + (size_t)BATCH * SEQ * DA;      // 2 MB
  __bf16* Vt  = Kb  + (size_t)BATCH * SEQ * DA;      // 4 MB
  __bf16* xbb = Vt  + (size_t)BATCH * SEQ * DV;      // 4 MB
  __bf16* Wqb = xbb + nX;
  __bf16* Wkb = Wqb + nWq;
  __bf16* Wvb = Wkb + nWk;
  float*  out = (float*)d_out;

  cvt_bf16_kernel<<<(int)(nX  / 1024), 256, 0, stream>>>(x,  xbb, (int)nX);
  cvt_bf16_kernel<<<(int)((nWq + 1023) / 1024), 256, 0, stream>>>(Wq, Wqb, (int)nWq);
  cvt_bf16_kernel<<<(int)((nWk + 1023) / 1024), 256, 0, stream>>>(Wk, Wkb, (int)nWk);
  cvt_bf16_kernel<<<(int)((nWv + 1023) / 1024), 256, 0, stream>>>(Wv, Wvb, (int)nWv);

  qkv_proj_wmma<<<(BATCH * SEQ) / 128, 256, 0, stream>>>(xbb, Wqb, Wkb, Wvb,
                                                         Qb, Kb, Vt);
  attn_flash_kernel<<<dim3(SEQ / 128, BATCH), 256, 0, stream>>>(Qb, Kb, Vt, out);
}
